// MoEDecoder_55997783605676
// MI455X (gfx1250) — compile-verified, matrix operand order checked by reading, not run on hardware
//
#include <hip/hip_runtime.h>
#include <math.h>

typedef _Float16 v16h __attribute__((ext_vector_type(16)));
typedef _Float16 v8h  __attribute__((ext_vector_type(8)));
typedef float    v8f  __attribute__((ext_vector_type(8)));
typedef int      v4i  __attribute__((ext_vector_type(4)));

#define AS1 __attribute__((address_space(1)))
#define AS3 __attribute__((address_space(3)))

#define L_   2
#define D_   1024
#define H_   16
#define HD_  64
#define V_   50257
#define Tm_  2048
#define CD_  512
#define E_   8
#define F_   4096
#define B_   2
#define BT_  4096
#define EPS_ 1e-5f

// ---------------------------------------------------------------------------
// Fragment loader: 16-bit A/B operand layout (ISA 7.12.2).
// Per lane: row fixed, two contiguous 8-half chunks at k = kbase + hlf*8 and
// kbase + 16 + hlf*8  ->  two ds_load_b128.
// ---------------------------------------------------------------------------
__device__ __forceinline__ v16h ld_frag(const _Float16* rowptr, int kbase, int hlf) {
    union U { v16h v; v8h h[2]; } u;
    u.h[0] = *(const v8h*)(rowptr + kbase + hlf * 8);
    u.h[1] = *(const v8h*)(rowptr + kbase + 16 + hlf * 8);
    return u.v;
}

__device__ __forceinline__ v8f zero8() {
    v8f z;
    for (int i = 0; i < 8; ++i) z[i] = 0.0f;
    return z;
}

// async memory->LDS copy of 16 bytes, with plain-load fallback
__device__ __forceinline__ void async_cp16(const _Float16* gsrc, _Float16* ldst) {
#if __has_builtin(__builtin_amdgcn_global_load_async_to_lds_b128)
    __builtin_amdgcn_global_load_async_to_lds_b128(
        (AS1 v4i*)(uintptr_t)gsrc,
        (AS3 v4i*)(unsigned)(uintptr_t)ldst, 0, 0);
#else
    *(v8h*)ldst = *(const v8h*)gsrc;
#endif
}
__device__ __forceinline__ void async_fence() {
#if __has_builtin(__builtin_amdgcn_global_load_async_to_lds_b128)
    asm volatile("s_wait_asynccnt 0" ::: "memory");
#endif
}

// ---------------------------------------------------------------------------
// Generic GEMM: C[M,N] = A[M,K] * W[N,K]^T (+ bias), f16 WMMA, f32 accum.
// Tile 128x128 (8 waves; wave = 32x64 = 2x4 WMMA tiles), K-step 32,
// double-buffered LDS; grid.x = M-tiles so co-scheduled blocks share W tiles
// in L2 (weights stream from HBM once; activations stay L2-resident).
// Compile-time specialization keeps the epilogue branch-free (no spills):
//   EPI: 0 store, 1 exact GELU, 2 scale by scale_ptr[scale_off]
//   OUT16: f16 output (ch) instead of fp32 C; BETA: accumulate into C;
//   USE_SEL: expert indirection; HAS_BIAS: add bias row.
// ---------------------------------------------------------------------------
#define LDS_STR 40   // 32 + 8 pad: 16B-aligned chunks, no bank conflicts

template <int EPI, bool OUT16, bool BETA, bool USE_SEL, bool HAS_BIAS>
__global__ __launch_bounds__(256)
void gemm_wmma(const float* __restrict__ A, const float* __restrict__ Wb,
               const int* __restrict__ sel, int sel_off, long long wstride,
               const float* __restrict__ biasb, long long bstride,
               const float* __restrict__ scale_ptr, int scale_off,
               float* __restrict__ C, _Float16* __restrict__ ch,
               int N, int K)
{
    __shared__ _Float16 As[2][128 * LDS_STR];
    __shared__ _Float16 Bs[2][128 * LDS_STR];

    const float* W    = Wb;
    const float* bias = biasb;
    if (USE_SEL) {
        int e = sel[sel_off];
        W += (long long)e * wstride;
        bias += (long long)e * bstride;
    }
    const int m0   = blockIdx.x * 128;
    const int n0   = blockIdx.y * 128;
    const int tid  = threadIdx.x;
    const int lane = tid & 31;
    const int wv   = tid >> 5;
    const int wm   = (wv & 3) * 32;
    const int wn   = (wv >> 2) * 64;
    const int r    = lane & 15;
    const int hlf  = lane >> 4;

    v8f acc[2][4];
    for (int i = 0; i < 2; ++i)
        for (int j = 0; j < 4; ++j) acc[i][j] = zero8();

    float4 ra[4], rb[4];
    auto ldregs = [&](int kt) {
        const int k0 = kt << 5;
        for (int it = 0; it < 4; ++it) {
            int idx = tid + it * 256;          // 0..1023
            int rr  = idx >> 3;
            int cc  = (idx & 7) * 4;
            ra[it] = *(const float4*)(A + (size_t)(m0 + rr) * K + k0 + cc);
            int n = n0 + rr;
            rb[it] = make_float4(0.f, 0.f, 0.f, 0.f);
            if (n < N) rb[it] = *(const float4*)(W + (size_t)n * K + k0 + cc);
        }
    };
    auto stlds = [&](int buf) {
        for (int it = 0; it < 4; ++it) {
            int idx = tid + it * 256;
            int rr  = idx >> 3;
            int cc  = (idx & 7) * 4;
            _Float16* da = &As[buf][rr * LDS_STR + cc];
            da[0] = (_Float16)ra[it].x; da[1] = (_Float16)ra[it].y;
            da[2] = (_Float16)ra[it].z; da[3] = (_Float16)ra[it].w;
            _Float16* db = &Bs[buf][rr * LDS_STR + cc];
            db[0] = (_Float16)rb[it].x; db[1] = (_Float16)rb[it].y;
            db[2] = (_Float16)rb[it].z; db[3] = (_Float16)rb[it].w;
        }
    };

    const int nk = K >> 5;
    ldregs(0);
    stlds(0);
    int cur = 0;
    for (int kt = 0; kt < nk; ++kt) {
        __syncthreads();
        const bool more = (kt + 1) < nk;
        if (more) ldregs(kt + 1);      // global loads in flight during WMMAs

        v16h af[2], bf[4];
        for (int mt = 0; mt < 2; ++mt)
            af[mt] = ld_frag(&As[cur][(wm + mt * 16 + r) * LDS_STR], 0, hlf);
        for (int nt = 0; nt < 4; ++nt)
            bf[nt] = ld_frag(&Bs[cur][(wn + nt * 16 + r) * LDS_STR], 0, hlf);
        for (int mt = 0; mt < 2; ++mt)
            for (int nt = 0; nt < 4; ++nt)
                acc[mt][nt] = __builtin_amdgcn_wmma_f32_16x16x32_f16(
                    false, af[mt], false, bf[nt], (short)0, acc[mt][nt], false, false);

        if (more) stlds(cur ^ 1);
        cur ^= 1;
    }

    float scl = 1.0f;
    if (EPI == 2) scl = scale_ptr[scale_off];
    for (int mt = 0; mt < 2; ++mt)
        for (int nt = 0; nt < 4; ++nt) {
            const int n = n0 + wn + nt * 16 + r;
            if (n >= N) continue;                 // only head GEMM edge tiles
            const int   mb = m0 + wm + mt * 16 + hlf * 8;
            const float bv = HAS_BIAS ? bias[n] : 0.0f;
            for (int v = 0; v < 8; ++v) {
                float val = acc[mt][nt][v] + bv;
                if (EPI == 1) val = 0.5f * val * (1.0f + erff(val * 0.70710678118654752f));
                if (EPI == 2) val *= scl;
                if (OUT16) {
                    ch[(size_t)(mb + v) * N + n] = (_Float16)val;
                } else {
                    float* cp = C + (size_t)(mb + v) * N + n;
                    if (BETA) val += *cp;
                    *cp = val;
                }
            }
        }
}

// ---------------------------------------------------------------------------
// Flash attention over f16 qkv: block = (q-tile 128) x (b,h); 8 waves x 16 q
// rows, streaming 64-key chunks. Q/K tiles staged via async mem->LDS copies
// (ASYNCcnt), V staged transposed so P*V B-fragments are contiguous.
// ---------------------------------------------------------------------------
#define FA_STR 72   // 64 + 8 pad

__global__ __launch_bounds__(256)
void flash_attn(const _Float16* __restrict__ qkv, float* __restrict__ out)
{
    __shared__ _Float16 Qs[128 * FA_STR];
    __shared__ _Float16 Ks[64 * FA_STR];
    __shared__ _Float16 Vt[HD_ * FA_STR];          // transposed: [d][key]
    __shared__ _Float16 Ps[8 * 16 * FA_STR];       // per-wave P staging

    const int qt  = blockIdx.x;
    const int bh  = blockIdx.y;
    const int bb  = bh / H_;
    const int hh  = bh % H_;
    const int tid = threadIdx.x;
    const int lane = tid & 31;
    const int wv   = tid >> 5;
    const int r    = lane & 15;
    const int hlf  = lane >> 4;

    const size_t rowstr = (size_t)3 * D_;
    const _Float16* qbase = qkv + (size_t)bb * Tm_ * rowstr + hh * HD_;
    const _Float16* kbase = qbase + D_;
    const _Float16* vbase = qbase + 2 * D_;

    // Q tile (128 x 64 f16): async copy, 16B per lane
    for (int idx = tid; idx < 128 * 8; idx += 256) {
        int rr = idx >> 3, cc = (idx & 7) * 8;
        async_cp16(qbase + (size_t)(qt * 128 + rr) * rowstr + cc,
                   Qs + rr * FA_STR + cc);
    }
    async_fence();
    __syncthreads();

    v16h qa[2];
    qa[0] = ld_frag(Qs + (wv * 16 + r) * FA_STR, 0,  hlf);
    qa[1] = ld_frag(Qs + (wv * 16 + r) * FA_STR, 32, hlf);

    v8f O[4];
    for (int i = 0; i < 4; ++i) O[i] = zero8();
    float mrow[8], lrow[8];
    for (int v = 0; v < 8; ++v) { mrow[v] = -__builtin_inff(); lrow[v] = 0.0f; }

    const int nkc = (qt + 1) * 2;  // 64-key chunks through the diagonal
    for (int kc = 0; kc < nkc; ++kc) {
        __syncthreads();
        // K chunk (64 x 64): async copy
        for (int idx = tid; idx < 64 * 8; idx += 256) {
            int rr = idx >> 3, cc = (idx & 7) * 8;
            async_cp16(kbase + (size_t)(kc * 64 + rr) * rowstr + cc,
                       Ks + rr * FA_STR + cc);
        }
        // V chunk transposed: Vt[d][key]
        for (int idx = tid; idx < 64 * 8; idx += 256) {
            int rr = idx >> 3, cc = (idx & 7) * 8;   // rr = key, cc = d
            v8h f = *(const v8h*)(vbase + (size_t)(kc * 64 + rr) * rowstr + cc);
            for (int i = 0; i < 8; ++i) Vt[(cc + i) * FA_STR + rr] = f[i];
        }
        async_fence();
        __syncthreads();

        // S = Q K^T * 1/8, causal-masked
        v8f s[4];
        for (int nt = 0; nt < 4; ++nt) {
            v8f a = zero8();
            for (int ks = 0; ks < 2; ++ks) {
                v16h bk = ld_frag(Ks + (nt * 16 + r) * FA_STR, ks * 32, hlf);
                a = __builtin_amdgcn_wmma_f32_16x16x32_f16(
                        false, qa[ks], false, bk, (short)0, a, false, false);
            }
            s[nt] = a;
        }
        for (int nt = 0; nt < 4; ++nt)
            for (int v = 0; v < 8; ++v) {
                float sv = s[nt][v] * 0.125f;
                int ng = kc * 64 + nt * 16 + r;
                int mg = qt * 128 + wv * 16 + v + hlf * 8;
                if (ng > mg) sv = -__builtin_inff();
                s[nt][v] = sv;
            }

        // online softmax (rows live across the 16 lanes of a half)
        for (int v = 0; v < 8; ++v) {
            float mx = -__builtin_inff();
            for (int nt = 0; nt < 4; ++nt) mx = fmaxf(mx, s[nt][v]);
            for (int msk = 8; msk >= 1; msk >>= 1)
                mx = fmaxf(mx, __shfl_xor(mx, msk, 32));
            float mnew  = fmaxf(mrow[v], mx);
            float alpha = expf(mrow[v] - mnew);
            mrow[v] = mnew;
            lrow[v] *= alpha;
            for (int ot = 0; ot < 4; ++ot) O[ot][v] *= alpha;
            float ps = 0.0f;
            for (int nt = 0; nt < 4; ++nt) {
                float p = expf(s[nt][v] - mnew);
                Ps[(wv * 16 + v + hlf * 8) * FA_STR + nt * 16 + r] = (_Float16)p;
                ps += p;
            }
            for (int msk = 8; msk >= 1; msk >>= 1)
                ps += __shfl_xor(ps, msk, 32);
            lrow[v] += ps;
        }

        // O += P V
        for (int ot = 0; ot < 4; ++ot)
            for (int ks = 0; ks < 2; ++ks) {
                v16h pa = ld_frag(Ps + (wv * 16 + r) * FA_STR, ks * 32, hlf);
                v16h vb = ld_frag(Vt + (ot * 16 + r) * FA_STR, ks * 32, hlf);
                O[ot] = __builtin_amdgcn_wmma_f32_16x16x32_f16(
                            false, pa, false, vb, (short)0, O[ot], false, false);
            }
    }

    for (int ot = 0; ot < 4; ++ot)
        for (int v = 0; v < 8; ++v) {
            int mg = qt * 128 + wv * 16 + v + hlf * 8;
            out[((size_t)bb * Tm_ + mg) * D_ + hh * HD_ + ot * 16 + r] = O[ot][v] / lrow[v];
        }
}

// ---------------------------------------------------------------------------
// Router: c_mean over 64 states, softmax over E=8 logits, top-2 -> sel/wts.
// ---------------------------------------------------------------------------
__global__ void router_kernel(const float* __restrict__ c_states,
                              const float* __restrict__ router_w,
                              const float* __restrict__ router_b,
                              int* __restrict__ sel, float* __restrict__ wts)
{
    const int l = blockIdx.x;
    __shared__ float cm[CD_];
    __shared__ float lg[E_];
    const int tid = threadIdx.x;
    for (int j = tid; j < CD_; j += blockDim.x) {
        float s = 0.0f;
        for (int i = 0; i < 64; ++i) s += c_states[i * CD_ + j];
        cm[j] = s * (1.0f / 64.0f);
    }
    __syncthreads();
    if (tid < E_) {
        const float* w = router_w + ((size_t)l * E_ + tid) * CD_;
        float s = router_b[l * E_ + tid];
        for (int j = 0; j < CD_; ++j) s += w[j] * cm[j];
        lg[tid] = s;
    }
    __syncthreads();
    if (tid == 0) {
        float mx = lg[0];
        for (int e = 1; e < E_; ++e) mx = fmaxf(mx, lg[e]);
        float ex[E_];
        for (int e = 0; e < E_; ++e) ex[e] = expf(lg[e] - mx);
        int i1 = 0; float v1 = -1.0f;
        for (int e = 0; e < E_; ++e) if (ex[e] > v1) { v1 = ex[e]; i1 = e; }
        int i2 = 0; float v2 = -1.0f;
        for (int e = 0; e < E_; ++e) if (e != i1 && ex[e] > v2) { v2 = ex[e]; i2 = e; }
        float inv = 1.0f / (v1 + v2);
        sel[2 * l] = i1; sel[2 * l + 1] = i2;
        wts[2 * l] = v1 * inv; wts[2 * l + 1] = v2 * inv;
    }
}

// ---------------------------------------------------------------------------
// LayerNorm( x + res ) * g + b, in place on x. res may be null.
// ---------------------------------------------------------------------------
__global__ __launch_bounds__(256)
void ln_residual(float* __restrict__ x, const float* __restrict__ res,
                 const float* __restrict__ g, const float* __restrict__ bta)
{
    const int row = blockIdx.x, tid = threadIdx.x;
    __shared__ float red[256];
    float* xr = x + (size_t)row * D_;
    const float* rr = res ? res + (size_t)row * D_ : nullptr;
    float vals[4];
    float s = 0.0f;
    for (int i = 0; i < 4; ++i) {
        float v = xr[tid + i * 256];
        if (rr) v += rr[tid + i * 256];
        vals[i] = v; s += v;
    }
    red[tid] = s; __syncthreads();
    for (int off = 128; off; off >>= 1) {
        if (tid < off) red[tid] += red[tid + off];
        __syncthreads();
    }
    float mu = red[0] * (1.0f / D_);
    __syncthreads();
    float s2 = 0.0f;
    for (int i = 0; i < 4; ++i) { float d = vals[i] - mu; s2 += d * d; }
    red[tid] = s2; __syncthreads();
    for (int off = 128; off; off >>= 1) {
        if (tid < off) red[tid] += red[tid + off];
        __syncthreads();
    }
    float rs = rsqrtf(red[0] * (1.0f / D_) + EPS_);
    for (int i = 0; i < 4; ++i) {
        int c = tid + i * 256;
        xr[c] = (vals[i] - mu) * rs * g[c] + bta[c];
    }
}

// ---------------------------------------------------------------------------
// x[row] = embed_w[token] + pos_w[t]
// ---------------------------------------------------------------------------
__global__ void embed_kernel(const int* __restrict__ tokens,
                             const float* __restrict__ embed_w,
                             const float* __restrict__ pos_w,
                             float* __restrict__ x)
{
    const int row = blockIdx.x;
    const int t   = row % Tm_;
    const int tok = tokens[row];
    const float* e = embed_w + (size_t)tok * D_;
    const float* p = pos_w + (size_t)t * D_;
    float* xr = x + (size_t)row * D_;
    for (int i = threadIdx.x; i < D_; i += blockDim.x) xr[i] = e[i] + p[i];
}

// ---------------------------------------------------------------------------
extern "C" void kernel_launch(void* const* d_in, const int* in_sizes, int n_in,
                              void* d_out, int out_size, void* d_ws, size_t ws_size,
                              hipStream_t stream)
{
    (void)in_sizes; (void)n_in; (void)out_size; (void)ws_size;
    const int*   tokens   = (const int*)d_in[0];
    const float* c_states = (const float*)d_in[1];
    const float* embed_w  = (const float*)d_in[2];
    const float* pos_w    = (const float*)d_in[3];
    const float* in_w     = (const float*)d_in[4];
    const float* in_b     = (const float*)d_in[5];
    const float* out_w    = (const float*)d_in[6];
    const float* out_b    = (const float*)d_in[7];
    const float* ln_a_g   = (const float*)d_in[8];
    const float* ln_a_b   = (const float*)d_in[9];
    const float* router_w = (const float*)d_in[10];
    const float* router_b = (const float*)d_in[11];
    const float* e_w1     = (const float*)d_in[12];
    const float* e_b1     = (const float*)d_in[13];
    const float* e_w2     = (const float*)d_in[14];
    const float* e_b2     = (const float*)d_in[15];
    const float* ln_m_g   = (const float*)d_in[16];
    const float* ln_m_b   = (const float*)d_in[17];
    const float* ln_f_g   = (const float*)d_in[18];
    const float* ln_f_b   = (const float*)d_in[19];
    const float* head_w   = (const float*)d_in[20];

    char* p = (char*)d_ws;
    float*    x     = (float*)p;    p += (size_t)BT_ * D_ * 4;       // 16 MB
    _Float16* qkv16 = (_Float16*)p; p += (size_t)BT_ * 3 * D_ * 2;   // 24 MB
    float*    attn  = (float*)p;    p += (size_t)BT_ * D_ * 4;       // 16 MB
    float*    proj  = (float*)p;    p += (size_t)BT_ * D_ * 4;       // 16 MB
    float*    hbuf  = (float*)p;    p += (size_t)BT_ * F_ * 4;       // 64 MB
    float*    ybuf  = (float*)p;    p += (size_t)BT_ * D_ * 4;       // 16 MB
    int*      sel   = (int*)p;      p += 4 * sizeof(int);
    float*    wts   = (float*)p;

    embed_kernel<<<BT_, 256, 0, stream>>>(tokens, embed_w, pos_w, x);
    router_kernel<<<L_, 256, 0, stream>>>(c_states, router_w, router_b, sel, wts);

    for (int l = 0; l < L_; ++l) {
        // qkv projection -> f16 (consumed only by attention)
        gemm_wmma<0, true, false, false, true>
            <<<dim3(BT_ / 128, 3 * D_ / 128), 256, 0, stream>>>(
            x, in_w + (size_t)l * 3 * D_ * D_, nullptr, 0, 0,
            in_b + (size_t)l * 3 * D_, 0, nullptr, 0,
            nullptr, qkv16, 3 * D_, D_);
        // causal attention
        flash_attn<<<dim3(Tm_ / 128, B_ * H_), 256, 0, stream>>>(qkv16, attn);
        // output projection
        gemm_wmma<0, false, false, false, true>
            <<<dim3(BT_ / 128, D_ / 128), 256, 0, stream>>>(
            attn, out_w + (size_t)l * D_ * D_, nullptr, 0, 0,
            out_b + (size_t)l * D_, 0, nullptr, 0,
            proj, nullptr, D_, D_);
        ln_residual<<<BT_, 256, 0, stream>>>(x, proj, ln_a_g + l * D_, ln_a_b + l * D_);
        // MoE: two selected experts
        for (int k = 0; k < 2; ++k) {
            gemm_wmma<1, false, false, true, true>
                <<<dim3(BT_ / 128, F_ / 128), 256, 0, stream>>>(
                x, e_w1 + (size_t)l * E_ * F_ * D_, sel, 2 * l + k, (long long)F_ * D_,
                e_b1 + (size_t)l * E_ * F_, (long long)F_, nullptr, 0,
                hbuf, nullptr, F_, D_);
            if (k == 0)
                gemm_wmma<2, false, false, true, true>
                    <<<dim3(BT_ / 128, D_ / 128), 256, 0, stream>>>(
                    hbuf, e_w2 + (size_t)l * E_ * D_ * F_, sel, 2 * l + k, (long long)D_ * F_,
                    e_b2 + (size_t)l * E_ * D_, (long long)D_, wts, 2 * l + k,
                    ybuf, nullptr, D_, F_);
            else
                gemm_wmma<2, false, true, true, true>
                    <<<dim3(BT_ / 128, D_ / 128), 256, 0, stream>>>(
                    hbuf, e_w2 + (size_t)l * E_ * D_ * F_, sel, 2 * l + k, (long long)D_ * F_,
                    e_b2 + (size_t)l * E_ * D_, (long long)D_, wts, 2 * l + k,
                    ybuf, nullptr, D_, F_);
        }
        ln_residual<<<BT_, 256, 0, stream>>>(x, ybuf, ln_m_g + l * D_, ln_m_b + l * D_);
    }
    ln_residual<<<BT_, 256, 0, stream>>>(x, nullptr, ln_f_g, ln_f_b);
    // head GEMM -> logits (no bias)
    gemm_wmma<0, false, false, false, false>
        <<<dim3(BT_ / 128, (V_ + 127) / 128), 256, 0, stream>>>(
        x, head_w, nullptr, 0, 0, nullptr, 0, nullptr, 0,
        (float*)d_out, nullptr, V_, D_);
}
